// MultiHeadAttention_76605036691500
// MI455X (gfx1250) — matrix operand order repaired and checked by reading, not compiled, not run on hardware
//
#include <hip/hip_runtime.h>
#include <hip/hip_bf16.h>

#define B_  2
#define S_  4096
#define D_  768
#define H_  12
#define DK_ 64
#define PSTR 4104   // f16 elements per LDS score row (4096 + 8 pad, keeps 16B align)

typedef __attribute__((ext_vector_type(16))) _Float16 v16h;
typedef __attribute__((ext_vector_type(8)))  float    v8f;

union FragH { v16h v; _Float16 h[16]; };

__device__ __forceinline__ v8f wmma16x16x32(v16h a, v16h b, v8f c) {
  // (neg_a, A, neg_b, B, c_mod, C, reuse_a, reuse_b)
  return __builtin_amdgcn_wmma_f32_16x16x32_f16(false, a, false, b, (short)0, c,
                                                false, false);
}

// ---------------------------------------------------------------------------
// Kernel 1: projection GEMM.  Out[m,n] = sum_k X[m,k]*W[n,k] + bias[n]
// X: [8192 x 768] f32, W: [768 x 768] f32.  Result f16.
//   TR=false: head-split  Out[((b*H+h)*S + s)*DK + dk]   (Q, K)
//   TR=true : transposed  Out[((b*H+h)*DK + dk)*S + s]   (V -> contiguous
//             K-dim for the P.V B-fragments)
// Wave tile: 16 rows x 64 cols.
// ---------------------------------------------------------------------------
template <bool TR>
__global__ __launch_bounds__(256) void proj_qkv(const float* __restrict__ X,
                                                const float* __restrict__ W,
                                                const float* __restrict__ bias,
                                                _Float16* __restrict__ Out) {
  const int lane = threadIdx.x & 31;
  const int wave = threadIdx.x >> 5;
  const int tile = blockIdx.x * 8 + wave;      // 0..6143
  const int rowBase = (tile / 12) * 16;
  const int colBase = (tile % 12) * 64;
  const int nLane = lane & 15;
  const int up = (lane & 16) ? 1 : 0;

  v8f acc[4] = {};
  const float* xrow = X + (size_t)(rowBase + nLane) * D_;

  for (int k0 = 0; k0 < D_; k0 += 32) {
    __builtin_prefetch(xrow + k0 + 64, 0, 1);
    FragH a;
    {
      const float* p = xrow + k0 + (up ? 16 : 0);
#pragma unroll
      for (int j = 0; j < 8; ++j) a.h[j] = (_Float16)p[j];
#pragma unroll
      for (int j = 0; j < 8; ++j) a.h[8 + j] = (_Float16)p[8 + j];
    }
#pragma unroll
    for (int t = 0; t < 4; ++t) {
      FragH b;
      const float* p = W + (size_t)(colBase + t * 16 + nLane) * D_ + k0 + (up ? 16 : 0);
#pragma unroll
      for (int j = 0; j < 16; ++j) b.h[j] = (_Float16)p[j];
      acc[t] = wmma16x16x32(a.v, b.v, acc[t]);
    }
  }

  const int rOff = up ? 8 : 0;
#pragma unroll
  for (int t = 0; t < 4; ++t) {
    const int n = colBase + t * 16 + nLane;
    const int h = n >> 6, dk = n & 63;
    const float bb = bias[n];
#pragma unroll
    for (int r = 0; r < 8; ++r) {
      const int mr = rowBase + rOff + r;
      const int bi = mr >> 12;
      const int ss = mr & (S_ - 1);
      const _Float16 val = (_Float16)(acc[t][r] + bb);
      if (TR)
        Out[(((size_t)(bi * H_ + h) * DK_) + dk) * S_ + ss] = val;
      else
        Out[(((size_t)(bi * H_ + h) * S_) + ss) * DK_ + dk] = val;
    }
  }
}

// ---------------------------------------------------------------------------
// Kernel 2 (fused): per block = one (bh, 16-row q tile):
//   P0: async-DMA the 16x64 Q tile to LDS (global_load_async_to_lds_b128,
//       drained with s_wait_asynccnt) -- shared by all 8 waves
//   P1: scores = 0.125 * Q K^T  -> LDS (f16), 8 waves split 4096 keys
//   P2: row max, exp-in-place, row sum (f32 math, LDS staging)
//   P3: normalize; write f32 attn weights to d_out (the only HBM-scale I/O)
//   P4: O = P @ V^T from LDS, split-K over waves, ds_add_f32 reduction
//   P5: store O as f16 head-split
// LDS total ~ 136 KB (< 320 KB WGP LDS)
// ---------------------------------------------------------------------------
__global__ __launch_bounds__(256) void attn_fused(const _Float16* __restrict__ Qh,
                                                  const _Float16* __restrict__ Kh,
                                                  const _Float16* __restrict__ Vt,
                                                  float* __restrict__ attnW,
                                                  _Float16* __restrict__ Oh) {
  __shared__ _Float16 sc[16 * PSTR];     // 131,328 B score/exp/weight staging
  __shared__ _Float16 qtile[16 * DK_];   // 2 KB async-staged Q tile
  __shared__ float obuf[16 * 64];        // 4 KB  P.V accumulator
  __shared__ float rowred[16][17];
  __shared__ float rowmax[16];
  __shared__ float rowinv[16];

  const int tid  = threadIdx.x;
  const int lane = tid & 31;
  const int wave = tid >> 5;
  const int bh = blockIdx.y;
  const int qBase = blockIdx.x * 16;
  const int up = (lane & 16) ? 1 : 0;
  const int nLane = lane & 15;
  const int rOff = up ? 8 : 0;

  const _Float16* Qb = Qh + (size_t)bh * S_ * DK_;
  const _Float16* Kb = Kh + (size_t)bh * S_ * DK_;
  const _Float16* Vb = Vt + (size_t)bh * DK_ * S_;   // [dk][s]
  float* Wrow = attnW + ((size_t)bh * S_ + qBase) * S_;

  // ---- P0: zero P.V accumulator + async Q-tile DMA to LDS ----
  for (int o = tid; o < 16 * 64; o += 256) obuf[o] = 0.f;
  if (tid < 128) {  // 128 lanes x 16B = 2 KB Q tile
    const unsigned long long ga =
        (unsigned long long)(Qb + (size_t)(qBase + (tid >> 3)) * DK_ + (tid & 7) * 8);
    const unsigned int lo = (unsigned int)(unsigned long long)(qtile + tid * 8);
    asm volatile("global_load_async_to_lds_b128 %0, %1, off"
                 :: "v"(lo), "v"(ga) : "memory");
  }
  asm volatile("s_wait_asynccnt 0x0" ::: "memory");
  __syncthreads();

  // ---- P1: scores into LDS ----
  FragH a0, a1;
  {
    const _Float16* p = qtile + nLane * DK_;
    const int off = up ? 16 : 0;
#pragma unroll
    for (int j = 0; j < 8; ++j) {
      a0.h[j]     = p[off + j];       a0.h[8 + j] = p[off + 8 + j];
      a1.h[j]     = p[32 + off + j];  a1.h[8 + j] = p[32 + off + 8 + j];
    }
  }
  const int kOffB = up ? 16 : 0;
  for (int g = 0; g < 8; ++g) {
    const int colBase = wave * 512 + g * 64;
#pragma unroll
    for (int t = 0; t < 4; ++t) {
      const int n = colBase + t * 16 + nLane;
      const _Float16* p = Kb + (size_t)n * DK_;
      FragH b0, b1;
#pragma unroll
      for (int j = 0; j < 16; ++j) b0.h[j] = p[kOffB + j];
#pragma unroll
      for (int j = 0; j < 16; ++j) b1.h[j] = p[32 + kOffB + j];
      v8f c = {};
      c = wmma16x16x32(a0.v, b0.v, c);
      c = wmma16x16x32(a1.v, b1.v, c);
#pragma unroll
      for (int r = 0; r < 8; ++r)
        sc[(rOff + r) * PSTR + n] = (_Float16)(c[r] * 0.125f);
    }
  }
  __syncthreads();

  // ---- P2a: row max (16 threads/row, 256-elem chunks) ----
  const int rr = tid & 15;
  const int ch = tid >> 4;
  {
    const _Float16* p = sc + rr * PSTR + ch * 256;
    float m = -3.4e38f;
#pragma unroll 8
    for (int i = 0; i < 256; ++i) m = fmaxf(m, (float)p[i]);
    rowred[rr][ch] = m;
  }
  __syncthreads();
  if (tid < 16) {
    float m = rowred[tid][0];
#pragma unroll
    for (int i = 1; i < 16; ++i) m = fmaxf(m, rowred[tid][i]);
    rowmax[tid] = m;
  }
  __syncthreads();

  // ---- P2b: exp in place + row sum ----
  {
    _Float16* p = sc + rr * PSTR + ch * 256;
    const float m = rowmax[rr];
    float s = 0.f;
#pragma unroll 8
    for (int i = 0; i < 256; ++i) {
      const float e = __expf((float)p[i] - m);
      s += e;
      p[i] = (_Float16)e;
    }
    rowred[rr][ch] = s;
  }
  __syncthreads();
  if (tid < 16) {
    float s = rowred[tid][0];
#pragma unroll
    for (int i = 1; i < 16; ++i) s += rowred[tid][i];
    rowinv[tid] = 1.0f / s;
  }
  __syncthreads();

  // ---- P3: normalize in LDS + coalesced f32 weight write to d_out ----
  for (int idx = tid; idx < 16 * 4096; idx += 256) {
    const int r = idx >> 12;
    const int col = idx & 4095;
    const float w = (float)sc[r * PSTR + col] * rowinv[r];
    sc[r * PSTR + col] = (_Float16)w;
    Wrow[(size_t)r * S_ + col] = w;
  }
  __syncthreads();

  // ---- P4: O = P @ V, split-K over waves (wave owns 512 keys) ----
  // B-fragment from transposed V: lane's column dk is a contiguous K-run.
  v8f acc[4] = {};
  const int aOff = up ? 16 : 0;
  const int bOff = up ? 16 : 0;
  const _Float16* arow = sc + nLane * PSTR;
  for (int ks = 0; ks < 16; ++ks) {
    const int k0 = wave * 512 + ks * 32;
    FragH a;
#pragma unroll
    for (int j = 0; j < 8; ++j) {
      a.h[j]     = arow[k0 + aOff + j];
      a.h[8 + j] = arow[k0 + aOff + 8 + j];
    }
#pragma unroll
    for (int t = 0; t < 4; ++t) {
      FragH b;
      const _Float16* vp = Vb + (size_t)(t * 16 + nLane) * S_ + k0 + bOff;
#pragma unroll
      for (int j = 0; j < 16; ++j) b.h[j] = vp[j];
      acc[t] = wmma16x16x32(a.v, b.v, acc[t]);
    }
  }
  // cross-wave reduction via LDS float atomics (ds_add_f32)
#pragma unroll
  for (int t = 0; t < 4; ++t) {
#pragma unroll
    for (int r = 0; r < 8; ++r)
      atomicAdd(&obuf[(rOff + r) * 64 + t * 16 + nLane], acc[t][r]);
  }
  __syncthreads();

  // ---- P5: store O (f16, head-split) ----
  for (int o = tid; o < 16 * 64; o += 256) {
    const int r = o >> 6, dk = o & 63;
    Oh[((size_t)bh * S_ + qBase + r) * DK_ + dk] = (_Float16)obuf[o];
  }
}

// ---------------------------------------------------------------------------
// Kernel 3: output projection.  out[m,n] = sum_k Ofl[m,k]*Wo[n,k] + bo[n]
// Ofl[m, h*64+dk] lives at Oh[((b*H+h)*S + s)*DK + dk], m = b*S+s.
// ---------------------------------------------------------------------------
__global__ __launch_bounds__(256) void out_proj(const _Float16* __restrict__ Oh,
                                                const float* __restrict__ Wo,
                                                const float* __restrict__ bo,
                                                float* __restrict__ Out) {
  const int lane = threadIdx.x & 31;
  const int wave = threadIdx.x >> 5;
  const int tile = blockIdx.x * 8 + wave;
  const int rowBase = (tile / 12) * 16;
  const int colBase = (tile % 12) * 64;
  const int nLane = lane & 15;
  const int up = (lane & 16) ? 1 : 0;

  const int mA = rowBase + nLane;
  const int bA = mA >> 12;
  const int sA = mA & (S_ - 1);
  const size_t rowb = (size_t)bA * H_ * S_ * DK_ + (size_t)sA * DK_;

  v8f acc[4] = {};
  for (int k0 = 0; k0 < D_; k0 += 32) {
    FragH a;
    {
      const int c0 = k0 + (up ? 16 : 0);
      const int c1 = c0 + 8;
      const _Float16* p0 = Oh + rowb + (size_t)(c0 >> 6) * ((size_t)S_ * DK_) + (c0 & 63);
      const _Float16* p1 = Oh + rowb + (size_t)(c1 >> 6) * ((size_t)S_ * DK_) + (c1 & 63);
#pragma unroll
      for (int j = 0; j < 8; ++j) { a.h[j] = p0[j]; a.h[8 + j] = p1[j]; }
    }
#pragma unroll
    for (int t = 0; t < 4; ++t) {
      FragH b;
      const float* p = Wo + (size_t)(colBase + t * 16 + nLane) * D_ + k0 + (up ? 16 : 0);
#pragma unroll
      for (int j = 0; j < 16; ++j) b.h[j] = (_Float16)p[j];
      acc[t] = wmma16x16x32(a.v, b.v, acc[t]);
    }
  }

  const int rOff = up ? 8 : 0;
#pragma unroll
  for (int t = 0; t < 4; ++t) {
    const int n = colBase + t * 16 + nLane;
    const float bb = bo[n];
#pragma unroll
    for (int r = 0; r < 8; ++r) {
      const int mr = rowBase + rOff + r;
      Out[(size_t)mr * D_ + n] = acc[t][r] + bb;
    }
  }
}

// ---------------------------------------------------------------------------
extern "C" void kernel_launch(void* const* d_in, const int* in_sizes, int n_in,
                              void* d_out, int out_size, void* d_ws, size_t ws_size,
                              hipStream_t stream) {
  (void)in_sizes; (void)n_in; (void)out_size; (void)ws_size;
  const float* query = (const float*)d_in[0];
  const float* key_  = (const float*)d_in[1];
  const float* value = (const float*)d_in[2];
  const float* Wq = (const float*)d_in[3];
  const float* bq = (const float*)d_in[4];
  const float* Wk = (const float*)d_in[5];
  const float* bk = (const float*)d_in[6];
  const float* Wv = (const float*)d_in[7];
  const float* bv = (const float*)d_in[8];
  const float* Wo = (const float*)d_in[9];
  const float* bo = (const float*)d_in[10];

  float* out   = (float*)d_out;                       // [B,S,D]
  float* attnW = out + (size_t)B_ * S_ * D_;          // [B,H,S,S]

  const size_t nH = (size_t)B_ * S_ * D_;             // == B*H*S*DK
  _Float16* Qh = (_Float16*)d_ws;
  _Float16* Kh = Qh + nH;
  _Float16* Vt = Kh + nH;                             // [B,H,DK,S]
  _Float16* Oh = Vt + nH;                             // total ~50 MB of ws

  dim3 blk(256);
  proj_qkv<false><<<768, blk, 0, stream>>>(query, Wq, bq, Qh);
  proj_qkv<false><<<768, blk, 0, stream>>>(key_,  Wk, bk, Kh);
  proj_qkv<true ><<<768, blk, 0, stream>>>(value, Wv, bv, Vt);
  attn_fused<<<dim3(S_ / 16, B_ * H_), blk, 0, stream>>>(Qh, Kh, Vt, attnW, Oh);
  out_proj<<<768, blk, 0, stream>>>(Oh, Wo, bo, out);
}
